// TGAN_4105988735525
// MI455X (gfx1250) — matrix operand order repaired and checked by reading, not compiled, not run on hardware
//
#include <hip/hip_runtime.h>
#include <hip/hip_bf16.h>

// ---- problem constants (B=4096, N=64, M=192, H=2, DK=96) ----
#define NSEQ   64
#define MDIM   192
#define KSTR   200   // f16 LDS row stride for K: 100 dwords -> conflict-free b128 reads
#define NEGSL  0.2f
#define MASKF  -1e10f

typedef __attribute__((ext_vector_type(16))) _Float16 v16h;
typedef __attribute__((ext_vector_type(8)))  _Float16 v8h;
typedef __attribute__((ext_vector_type(8)))  float    v8f;

// ---------------------------------------------------------------------------
// Prep kernel: fold Wq/Wk through w_map into 2x192 score maps, convert Wv->f16.
//   maps[0..383]   = wqmap[h][m] = sum_d wq[h*96+d][m] * w_map[d]
//   maps[384..767] = wkmap[h][m] = sum_d wk[h*96+d][m] * w_map[96+d]
// ---------------------------------------------------------------------------
__global__ __launch_bounds__(256) void tgat_prep(
    const float* __restrict__ wq, const float* __restrict__ wk,
    const float* __restrict__ wv, const float* __restrict__ w_map,
    float* __restrict__ maps, _Float16* __restrict__ wv16) {
  int idx = blockIdx.x * blockDim.x + threadIdx.x;
  if (idx < MDIM * MDIM) wv16[idx] = (_Float16)wv[idx];
  if (idx < 2 * 2 * MDIM) {
    int which = idx / (2 * MDIM);          // 0 = wqmap, 1 = wkmap
    int rem   = idx % (2 * MDIM);
    int h = rem / MDIM, m = rem % MDIM;
    const float* w  = which ? wk : wq;
    const float* wm = which ? (w_map + 96) : w_map;
    float acc = 0.f;
    #pragma unroll 4
    for (int d = 0; d < 96; ++d) acc += w[(h * 96 + d) * MDIM + m] * wm[d];
    maps[idx] = acc;
  }
}

// ---------------------------------------------------------------------------
// Main kernel: one workgroup (256 thr / 8 wave32) per batch row.
// ---------------------------------------------------------------------------
__global__ __launch_bounds__(256) void tgat_main(
    const float* __restrict__ src,  const float* __restrict__ src_t,
    const float* __restrict__ seq,  const float* __restrict__ seq_t,
    const float* __restrict__ seq_e, const unsigned char* __restrict__ mask,
    const float* __restrict__ maps, const _Float16* __restrict__ wv16,
    const float* __restrict__ fc_w, const float* __restrict__ fc_b,
    const float* __restrict__ ln_g, const float* __restrict__ ln_b,
    const float* __restrict__ m1_w, const float* __restrict__ m1_b,
    const float* __restrict__ m2_w, const float* __restrict__ m2_b,
    float* __restrict__ out_final, float* __restrict__ out_attn, int Bn) {

  __shared__ __align__(16) _Float16 sK[NSEQ * KSTR];  // K = [seq|seq_e|seq_t] as f16
  __shared__ __align__(32) _Float16 sA[16 * KSTR];    // cK padded to 16 rows (WMMA A)
  __shared__ float sQ[MDIM];                          // q = [src|0|src_t] (residual)
  __shared__ float sAttn[2 * NSEQ];
  __shared__ float sAo[MDIM];                         // attention output (pre-fc)
  __shared__ float sZ[MDIM];                          // fc + residual (LN input)
  __shared__ float sX[MDIM + 64];                     // [LN out | src] for MLP
  __shared__ float sH[64];
  __shared__ float sRed[4];                           // mu, rstd, sq[0], sq[1]

  const int b    = blockIdx.x;
  const int t    = threadIdx.x;
  const int lane = t & 31;
  const int wave = t >> 5;

  // ---- Phase 1: stage q (f32) and K (f16) into LDS; zero pad rows of sA ----
  if (t < 64) {
    sQ[t]       = src[(size_t)b * 64 + t];
    sQ[64 + t]  = 0.f;
    sQ[128 + t] = src_t[(size_t)b * 64 + t];
  }
  #pragma unroll
  for (int s = 0; s < 3; ++s) {
    const float* sp = (s == 0) ? seq : (s == 1 ? seq_e : seq_t);
    const int base = s * 64;
    #pragma unroll
    for (int j = 0; j < 4; ++j) {
      int f = t + j * 256;                  // float4 index 0..1023
      int n = f >> 4, col = (f & 15) << 2;
      float4 v = *(const float4*)(sp + (((size_t)b * 64 + n) << 6) + col);
      _Float16* d = sK + n * KSTR + base + col;
      d[0] = (_Float16)v.x; d[1] = (_Float16)v.y;
      d[2] = (_Float16)v.z; d[3] = (_Float16)v.w;
    }
  }
  // zero rows 2..15 of the padded A tile (rows 0,1 filled with cK later)
  for (int r = 2; r < 16; ++r) sA[r * KSTR + t % MDIM] = (_Float16)0.f;
  if (t < 2 * (KSTR - MDIM) * 16) { /* pad cols never read: k<192 only */ }
  __syncthreads();

  // ---- Phase 2: sq[h] = q . wqmap[h]  (waves 0,1) ----
  if (t < 64) {
    int h = wave;
    float acc = 0.f;
    for (int j = lane; j < MDIM; j += 32) acc += sQ[j] * maps[h * MDIM + j];
    for (int o = 16; o; o >>= 1) acc += __shfl_xor(acc, o);
    if (lane == 0) sRed[2 + h] = acc;
  }
  __syncthreads();

  // ---- Phase 3: sk[h][n] = K[n] . wkmap[h]; add sq, apply mask ----
  if (t < 128) {
    int n = t >> 1, h = t & 1;
    const float* mp = maps + 2 * MDIM + h * MDIM;
    const _Float16* kr = sK + n * KSTR;
    float acc = 0.f;
    #pragma unroll 4
    for (int m = 0; m < MDIM; ++m) acc += (float)kr[m] * mp[m];
    float a = sRed[2 + h] + acc;
    if (mask[(size_t)b * 64 + n]) a = MASKF;
    sAttn[h * NSEQ + n] = a;
  }
  __syncthreads();

  // ---- Phase 4: softmax over n per head (wave h), emit attn output ----
  if (t < 64) {
    int h = wave;
    float v0 = sAttn[h * NSEQ + lane];
    float v1 = sAttn[h * NSEQ + 32 + lane];
    float mx = fmaxf(v0, v1);
    for (int o = 16; o; o >>= 1) mx = fmaxf(mx, __shfl_xor(mx, o));
    float e0 = __expf(v0 - mx), e1 = __expf(v1 - mx);
    float s = e0 + e1;
    for (int o = 16; o; o >>= 1) s += __shfl_xor(s, o);
    float inv = 1.f / s;
    e0 *= inv; e1 *= inv;
    sAttn[h * NSEQ + lane] = e0;
    sAttn[h * NSEQ + 32 + lane] = e1;
    float* ao = out_attn + ((size_t)h * Bn + b) * NSEQ;   // row h*B+b
    ao[lane] = e0; ao[lane + 32] = e1;
  }
  __syncthreads();

  // ---- Phase 5: cK[h][m] = sum_n attn[h][n]*K[n][m]  -> rows 0,1 of sA ----
  if (t < MDIM) {
    float a0 = 0.f, a1 = 0.f;
    #pragma unroll 4
    for (int n = 0; n < NSEQ; ++n) {
      float kv = (float)sK[n * KSTR + t];
      a0 += sAttn[n] * kv;
      a1 += sAttn[NSEQ + n] * kv;
    }
    sA[t]        = (_Float16)a0;
    sA[KSTR + t] = (_Float16)a1;
  }
  __syncthreads();

  // ---- Phase 6: out = cK_pad(16x192) @ Wv^T via v_wmma_f32_16x16x32_f16 ----
  // 12 output tiles of 16 cols, 6 K-steps of 32. Waves 0..7 take tiles w, w+8.
  {
    const int khalf = lane >> 4;            // K sub-group per documented A/B layout
    const int ocl   = lane & 15;
    #pragma unroll
    for (int rep = 0; rep < 2; ++rep) {
      int otile = wave + rep * 8;
      if (otile < 12) {
        int ocol = otile * 16 + ocl;
        v8f acc = {};
        #pragma unroll
        for (int ks = 0; ks < 6; ++ks) {
          // A fragment (LDS): lane row = lane&15; halves {k..k+7, k+16..k+23}
          const _Float16* ap = sA + ocl * KSTR + ks * 32 + khalf * 8;
          v8h a0 = *(const v8h*)ap;
          v8h a1 = *(const v8h*)(ap + 16);
          v16h af = __builtin_shufflevector(a0, a1, 0,1,2,3,4,5,6,7,
                                                    8,9,10,11,12,13,14,15);
          // B fragment (global, L2-hot): B[m][o]=Wv[o][m]; halves {k..k+7, k+16..k+23}
          const _Float16* bp = wv16 + (size_t)ocol * MDIM + ks * 32 + khalf * 8;
          v8h b0 = *(const v8h*)bp;
          v8h b1 = *(const v8h*)(bp + 16);
          v16h bf = __builtin_shufflevector(b0, b1, 0,1,2,3,4,5,6,7,
                                                    8,9,10,11,12,13,14,15);
          acc = __builtin_amdgcn_wmma_f32_16x16x32_f16(
              false, af, false, bf, (short)0, acc, false, false);
        }
        // D rows 0 (head 0) / 1 (head 1) live in lanes 0-15, VGPR 0/1
        if (khalf == 0) sAo[ocol] = (otile < 6) ? acc[0] : acc[1];
      }
    }
  }
  __syncthreads();

  // ---- Phase 7: fc + leaky_relu + residual ----
  if (t < MDIM) {
    const float* wrow = fc_w + (size_t)t * MDIM;
    float acc = fc_b[t];
    #pragma unroll 4
    for (int o = 0; o < MDIM; ++o) acc += sAo[o] * wrow[o];
    acc = (acc > 0.f) ? acc : NEGSL * acc;
    sZ[t] = acc + sQ[t];
  }
  __syncthreads();

  // ---- Phase 8: LayerNorm over 192 ----
  if (wave == 0) {
    float s = 0.f, s2 = 0.f;
    for (int j = lane; j < MDIM; j += 32) { float z = sZ[j]; s += z; s2 += z * z; }
    for (int o = 16; o; o >>= 1) { s += __shfl_xor(s, o); s2 += __shfl_xor(s2, o); }
    if (lane == 0) {
      float mu = s / (float)MDIM;
      float var = s2 / (float)MDIM - mu * mu;
      sRed[0] = mu;
      sRed[1] = rsqrtf(var + 1e-5f);
    }
  }
  __syncthreads();
  if (t < MDIM) sX[t] = (sZ[t] - sRed[0]) * sRed[1] * ln_g[t] + ln_b[t];
  if (t < 64)  sX[MDIM + t] = src[(size_t)b * 64 + t];
  __syncthreads();

  // ---- Phase 9: MLP ----
  if (t < 64) {
    const float* w1 = m1_w + (size_t)t * (MDIM + 64);
    float acc = m1_b[t];
    #pragma unroll 4
    for (int j = 0; j < MDIM + 64; ++j) acc += sX[j] * w1[j];
    sH[t] = fmaxf(acc, 0.f);
  }
  __syncthreads();
  if (t < 64) {
    const float* w2 = m2_w + (size_t)t * 64;
    float acc = m2_b[t];
    #pragma unroll 4
    for (int j = 0; j < 64; ++j) acc += sH[j] * w2[j];
    out_final[(size_t)b * 64 + t] = acc;
  }
}

// ---------------------------------------------------------------------------
extern "C" void kernel_launch(void* const* d_in, const int* in_sizes, int n_in,
                              void* d_out, int out_size, void* d_ws, size_t ws_size,
                              hipStream_t stream) {
  const float* src   = (const float*)d_in[0];
  const float* src_t = (const float*)d_in[1];
  const float* seq   = (const float*)d_in[2];
  const float* seq_t = (const float*)d_in[3];
  const float* seq_e = (const float*)d_in[4];
  const unsigned char* mask = (const unsigned char*)d_in[5];   // jax bool -> 1 byte
  const float* wq    = (const float*)d_in[6];
  const float* wk    = (const float*)d_in[7];
  const float* wv    = (const float*)d_in[8];
  const float* w_map = (const float*)d_in[9];
  const float* fc_w  = (const float*)d_in[10];
  const float* fc_b  = (const float*)d_in[11];
  const float* ln_g  = (const float*)d_in[12];
  const float* ln_b  = (const float*)d_in[13];
  const float* m1_w  = (const float*)d_in[14];
  const float* m1_b  = (const float*)d_in[15];
  const float* m2_w  = (const float*)d_in[16];
  const float* m2_b  = (const float*)d_in[17];

  const int Bn = in_sizes[0] / 64;                      // 4096

  float*    maps = (float*)d_ws;                        // 768 f32
  _Float16* wv16 = (_Float16*)((char*)d_ws + 3072);     // 36864 f16 (32B aligned)

  tgat_prep<<<(MDIM * MDIM + 255) / 256, 256, 0, stream>>>(wq, wk, wv, w_map,
                                                           maps, wv16);

  float* out_final = (float*)d_out;
  float* out_attn  = out_final + (size_t)Bn * 64;

  tgat_main<<<Bn, 256, 0, stream>>>(src, src_t, seq, seq_t, seq_e, mask,
                                    maps, wv16, fc_w, fc_b, ln_g, ln_b,
                                    m1_w, m1_b, m2_w, m2_b,
                                    out_final, out_attn, Bn);
}
